// GxE_InteractionGNN_78073915507271
// MI455X (gfx1250) — matrix operand order repaired and checked by reading, not compiled, not run on hardware
//
#include <hip/hip_runtime.h>
#include <hip/hip_bf16.h>

// ---------------- problem constants ----------------
#define N_CNT   50000
#define E_CNT   400000
#define ET_CNT  (E_CNT + N_CNT)     // edges + self loops = 450000
#define T_CNT   200000
#define H_CNT   4
#define C_CNT   128
#define HC_CNT  512                 // H*C
#define IN_CNT  128
#define L_CNT   3
#define NEG_SLOPE 0.2f

typedef __attribute__((ext_vector_type(16))) _Float16 v16h;
typedef __attribute__((ext_vector_type(8)))  float    v8f;

// ---------------- helpers ----------------
__device__ __forceinline__ unsigned f2ord(float f) {
    unsigned u = __float_as_uint(f);
    return (u & 0x80000000u) ? ~u : (u | 0x80000000u);
}
__device__ __forceinline__ float ord2f(unsigned u) {
    return (u & 0x80000000u) ? __uint_as_float(u & 0x7fffffffu)
                             : __uint_as_float(~u);
}
#define ORD_NEG_INF 0x007FFFFFu   // f2ord(-inf)

__device__ __forceinline__ void edge_sd(const int* __restrict__ ei, int e, int& s, int& d) {
    if (e < E_CNT) { s = ei[e]; d = ei[E_CNT + e]; }
    else           { s = e - E_CNT; d = e - E_CNT; }
}

// Load A fragment (16x32 f16, row-major src, given lane)
__device__ __forceinline__ v16h load_afrag(const _Float16* __restrict__ row_ptr) {
    // row_ptr already points at (row, k + ks*8)
    v16h a;
#pragma unroll
    for (int i = 0; i < 8; ++i) { a[i] = row_ptr[i]; a[8 + i] = row_ptr[16 + i]; }
    return a;
}
// Load B fragment (32x16 f16): K halves [0..8) and [16..24) from p per lane
__device__ __forceinline__ v16h load_bfrag(const _Float16* __restrict__ p) {
    v16h b;
#pragma unroll
    for (int i = 0; i < 8; ++i) { b[i] = p[i]; b[8 + i] = p[16 + i]; }
    return b;
}

// ---------------- utility kernels ----------------
__global__ void fill_f32_kernel(float* p, float v, long n) {
    long i = (long)blockIdx.x * blockDim.x + threadIdx.x;
    if (i < n) p[i] = v;
}
__global__ void fill_u32_kernel(unsigned* p, unsigned v, long n) {
    long i = (long)blockIdx.x * blockDim.x + threadIdx.x;
    if (i < n) p[i] = v;
}
__global__ void f32_to_f16_kernel(const float* __restrict__ in, _Float16* __restrict__ out, long n) {
    long i = (long)blockIdx.x * blockDim.x + threadIdx.x;
    if (i < n) out[i] = (_Float16)in[i];
}

// ---------------- GAT layer: h = x @ W^T (N x 512), f16 WMMA ----------------
// grid = (N/16, 4), block = 256 (8 waves). Wave w covers 16 cols, block covers 128 cols.
__global__ __launch_bounds__(256)
void gat_gemm_kernel(const _Float16* __restrict__ xH,   // N x 128
                     const _Float16* __restrict__ WH,   // 512 x 128 (row-major, this layer)
                     float* __restrict__ h)             // N x 512
{
    const int rowbase = blockIdx.x * 16;
    const int lane = threadIdx.x & 31;
    const int wave = threadIdx.x >> 5;
    const int colbase = blockIdx.y * 128 + wave * 16;
    const int rA = lane & 15, ks = lane >> 4;

    const _Float16* arow = xH + (size_t)(rowbase + rA) * IN_CNT;
    const _Float16* brow = WH + (size_t)(colbase + rA) * IN_CNT;

    v8f acc = {};
#pragma unroll
    for (int k = 0; k < IN_CNT; k += 32) {
        v16h a = load_afrag(arow + k + ks * 8);
        v16h b = load_bfrag(brow + k + ks * 8);
        acc = __builtin_amdgcn_wmma_f32_16x16x32_f16(false, a, false, b,
                                                     (short)0, acc, false, false);
    }
    const int m0 = rowbase + ((lane >= 16) ? 8 : 0);
    const int nc = colbase + (lane & 15);
#pragma unroll
    for (int i = 0; i < 8; ++i)
        h[(size_t)(m0 + i) * HC_CNT + nc] = acc[i];
}

// ---------------- alpha_s / alpha_d: one wave per (node, head) ----------------
__global__ __launch_bounds__(256)
void alpha_kernel(const float* __restrict__ h,
                  const float* __restrict__ att_s,   // H x C (this layer)
                  const float* __restrict__ att_d,
                  float* __restrict__ aS, float* __restrict__ aD)
{
    long gw = ((long)blockIdx.x * blockDim.x + threadIdx.x) >> 5;
    int lane = threadIdx.x & 31;
    if (gw >= (long)N_CNT * H_CNT) return;
    int n = (int)(gw >> 2), hd = (int)(gw & 3);
    const float4* hp = (const float4*)(h + (size_t)n * HC_CNT + hd * C_CNT);
    const float4* sp = (const float4*)(att_s + hd * C_CNT);
    const float4* dp = (const float4*)(att_d + hd * C_CNT);
    float4 hv = hp[lane], sv = sp[lane], dv = dp[lane];
    float s = hv.x * sv.x + hv.y * sv.y + hv.z * sv.z + hv.w * sv.w;
    float d = hv.x * dv.x + hv.y * dv.y + hv.z * dv.z + hv.w * dv.w;
#pragma unroll
    for (int m = 16; m; m >>= 1) { s += __shfl_xor(s, m, 32); d += __shfl_xor(d, m, 32); }
    if (lane == 0) { aS[gw] = s; aD[gw] = d; }
}

// ---------------- edge pass 1: leaky-relu logits + segment max ----------------
__global__ void edge_max_kernel(const int* __restrict__ ei,
                                const float* __restrict__ aS, const float* __restrict__ aD,
                                float* __restrict__ elr, unsigned* __restrict__ mmax)
{
    int e = blockIdx.x * blockDim.x + threadIdx.x;
    if (e >= ET_CNT) return;
    int s, d; edge_sd(ei, e, s, d);
#pragma unroll
    for (int hd = 0; hd < H_CNT; ++hd) {
        float v = aS[s * H_CNT + hd] + aD[d * H_CNT + hd];
        v = (v > 0.f) ? v : NEG_SLOPE * v;
        elr[(size_t)e * H_CNT + hd] = v;
        atomicMax(&mmax[d * H_CNT + hd], f2ord(v));
    }
}

// ---------------- edge pass 2: exp(e - max) + segment sum ----------------
__global__ void edge_exp_kernel(const int* __restrict__ ei,
                                float* __restrict__ eexp,            // in: elr, out: exp
                                const unsigned* __restrict__ mmax,
                                float* __restrict__ denom)
{
    int e = blockIdx.x * blockDim.x + threadIdx.x;
    if (e >= ET_CNT) return;
    int s, d; edge_sd(ei, e, s, d);
#pragma unroll
    for (int hd = 0; hd < H_CNT; ++hd) {
        float m = ord2f(mmax[d * H_CNT + hd]);
        float ex = expf(eexp[(size_t)e * H_CNT + hd] - m);
        eexp[(size_t)e * H_CNT + hd] = ex;
        atomicAdd(&denom[d * H_CNT + hd], ex);
    }
}

// ---------------- edge pass 3: weighted scatter-add of h[src] ----------------
// 2 edges per 256-thread block; each thread owns 4 contiguous channels (float4 load,
// 4 scalar f32 atomics -- the atomic count is inherent to f32 accumulation).
__global__ __launch_bounds__(256)
void edge_aggr_kernel(const int* __restrict__ ei,
                      const float* __restrict__ eexp, const float* __restrict__ denom,
                      const float* __restrict__ h, float* __restrict__ outagg)
{
    int e = blockIdx.x * 2 + (threadIdx.x >> 7);
    int t = threadIdx.x & 127;          // channel group
    int c4 = t * 4;                     // 0..508
    int hd = t >> 5;                    // head
    int s, d; edge_sd(ei, e, s, d);
    float coef = eexp[(size_t)e * H_CNT + hd] / denom[d * H_CNT + hd];
    float4 hv = *(const float4*)(h + (size_t)s * HC_CNT + c4);
    float* o = outagg + (size_t)d * HC_CNT + c4;
    atomicAdd(o + 0, coef * hv.x);
    atomicAdd(o + 1, coef * hv.y);
    atomicAdd(o + 2, coef * hv.z);
    atomicAdd(o + 3, coef * hv.w);
}

// ---------------- head mean + bias (+relu) -> f16 next-layer input ----------------
__global__ void mean_kernel(const float* __restrict__ outagg,
                            const float* __restrict__ bg,   // C (this layer)
                            _Float16* __restrict__ xnextH, int do_relu)
{
    long i = (long)blockIdx.x * blockDim.x + threadIdx.x;   // N*128 exact
    if (i >= (long)N_CNT * IN_CNT) return;
    int n = (int)(i >> 7), c = (int)(i & 127);
    const float* p = outagg + (size_t)n * HC_CNT + c;
    float v = 0.25f * (p[0] + p[128] + p[256] + p[384]) + bg[c];
    if (do_relu) v = fmaxf(v, 0.f);
    xnextH[i] = (_Float16)v;
}

// ---------------- fused triplet MLP: gather + 384->128 relu -> 128->64 relu -> 64->1 --
#define ZSTR  392   // 384 + 8  (bank-conflict padding)
#define Z1STR 136   // 128 + 8
#define Z2STR 72    // 64 + 8
__global__ __launch_bounds__(256)
void mlp_kernel(const _Float16* __restrict__ hH,     // N x 128, final node features
                const int* __restrict__ trips,       // T x 3
                const _Float16* __restrict__ W1, const float* __restrict__ b1,
                const _Float16* __restrict__ W2, const float* __restrict__ b2,
                const float* __restrict__ W3, const float* __restrict__ b3,
                float* __restrict__ out)
{
    __shared__ __align__(16) _Float16 zH[16 * ZSTR];
    __shared__ __align__(16) _Float16 z1H[16 * Z1STR];
    __shared__ __align__(16) _Float16 z2H[16 * Z2STR];

    const int rowbase = blockIdx.x * 16;
    const int tid = threadIdx.x, lane = tid & 31, wave = tid >> 5;
    const int rA = lane & 15, ks = lane >> 4;

    // gather concat(g, ev, tr) into LDS in 16-byte chunks:
    // 16 rows x 3 segments x 16 chunks (8 halves each) = 768 b128 copies
    for (int idx = tid; idx < 16 * 3 * 16; idx += 256) {
        int r   = idx >> 5 >> 1;            // idx / 48... careful: compute directly
        r       = idx / 48;
        int rem = idx - r * 48;
        int seg = rem >> 4;                 // 0..2
        int c8  = (rem & 15) * 8;           // 0,8,...,120 halves
        int node = trips[(rowbase + r) * 3 + seg];
        *(float4*)(&zH[r * ZSTR + seg * 128 + c8]) =
            *(const float4*)(&hH[(size_t)node * 128 + c8]);
    }
    __syncthreads();

    // ---- GEMM1: 16x384 @ 384x128, wave w -> 16 cols ----
    {
        const int colbase = wave * 16;
        const _Float16* brow = W1 + (size_t)(colbase + rA) * 384;
        v8f acc = {};
#pragma unroll
        for (int k = 0; k < 384; k += 32) {
            v16h a = load_afrag(&zH[rA * ZSTR + k + ks * 8]);
            v16h b = load_bfrag(brow + k + ks * 8);
            acc = __builtin_amdgcn_wmma_f32_16x16x32_f16(false, a, false, b,
                                                         (short)0, acc, false, false);
        }
        const int m0 = (lane >= 16) ? 8 : 0;
        const int nc = colbase + (lane & 15);
        const float bias = b1[nc];
#pragma unroll
        for (int i = 0; i < 8; ++i)
            z1H[(m0 + i) * Z1STR + nc] = (_Float16)fmaxf(acc[i] + bias, 0.f);
    }
    __syncthreads();

    // ---- GEMM2: 16x128 @ 128x64, waves 0..3 ----
    if (wave < 4) {
        const int colbase = wave * 16;
        const _Float16* brow = W2 + (size_t)(colbase + rA) * 128;
        v8f acc = {};
#pragma unroll
        for (int k = 0; k < 128; k += 32) {
            v16h a = load_afrag(&z1H[rA * Z1STR + k + ks * 8]);
            v16h b = load_bfrag(brow + k + ks * 8);
            acc = __builtin_amdgcn_wmma_f32_16x16x32_f16(false, a, false, b,
                                                         (short)0, acc, false, false);
        }
        const int m0 = (lane >= 16) ? 8 : 0;
        const int nc = colbase + (lane & 15);
        const float bias = b2[nc];
#pragma unroll
        for (int i = 0; i < 8; ++i)
            z2H[(m0 + i) * Z2STR + nc] = (_Float16)fmaxf(acc[i] + bias, 0.f);
    }
    __syncthreads();

    // ---- final 64 -> 1 : 16 threads per row ----
    {
        int r = tid >> 4, j = tid & 15;
        float s = 0.f;
#pragma unroll
        for (int i = 0; i < 4; ++i)
            s += (float)z2H[r * Z2STR + j * 4 + i] * W3[j * 4 + i];
#pragma unroll
        for (int m = 8; m; m >>= 1) s += __shfl_xor(s, m, 16);
        if (j == 0) out[rowbase + r] = s + b3[0];
    }
}

// ---------------- host launcher ----------------
extern "C" void kernel_launch(void* const* d_in, const int* in_sizes, int n_in,
                              void* d_out, int out_size, void* d_ws, size_t ws_size,
                              hipStream_t stream) {
    const float* x      = (const float*)d_in[0];
    const int*   ei     = (const int*)  d_in[1];   // (2, E)
    const int*   trips  = (const int*)  d_in[2];   // (T, 3)
    const float* Wg     = (const float*)d_in[3];   // (L, 512, 128)
    const float* att_s  = (const float*)d_in[4];   // (L, H, C)
    const float* att_d  = (const float*)d_in[5];
    const float* bg     = (const float*)d_in[6];   // (L, C)
    const float* Wm1    = (const float*)d_in[7];   // (128, 384)
    const float* bm1    = (const float*)d_in[8];
    const float* Wm2    = (const float*)d_in[9];   // (64, 128)
    const float* bm2    = (const float*)d_in[10];
    const float* Wm3    = (const float*)d_in[11];  // (1, 64)
    const float* bm3    = (const float*)d_in[12];
    float* out = (float*)d_out;

    // workspace carve-out (256B aligned slices)
    size_t off = 0;
    auto carve = [&](size_t bytes) -> char* {
        char* p = (char*)d_ws + off;
        off += (bytes + 255) & ~(size_t)255;
        return p;
    };
    float*     hbuf   = (float*)    carve((size_t)N_CNT * HC_CNT * 4);  // 102.4 MB
    float*     outagg = (float*)    carve((size_t)N_CNT * HC_CNT * 4);  // 102.4 MB
    float*     aS     = (float*)    carve((size_t)N_CNT * H_CNT * 4);
    float*     aD     = (float*)    carve((size_t)N_CNT * H_CNT * 4);
    unsigned*  mmax   = (unsigned*) carve((size_t)N_CNT * H_CNT * 4);
    float*     denom  = (float*)    carve((size_t)N_CNT * H_CNT * 4);
    float*     eexp   = (float*)    carve((size_t)ET_CNT * H_CNT * 4);  // 7.2 MB
    _Float16*  xH     = (_Float16*) carve((size_t)N_CNT * IN_CNT * 2);  // 12.8 MB
    _Float16*  hfinH  = (_Float16*) carve((size_t)N_CNT * IN_CNT * 2);  // 12.8 MB
    _Float16*  WgH    = (_Float16*) carve((size_t)L_CNT * HC_CNT * IN_CNT * 2);
    _Float16*  Wm1H   = (_Float16*) carve((size_t)128 * 384 * 2);
    _Float16*  Wm2H   = (_Float16*) carve((size_t)64 * 128 * 2);

    auto blocks = [](long n, int b) { return (unsigned)((n + b - 1) / b); };

    // precision staging: f16 copies of GEMM operands
    f32_to_f16_kernel<<<blocks((long)N_CNT * IN_CNT, 256), 256, 0, stream>>>(x, xH, (long)N_CNT * IN_CNT);
    f32_to_f16_kernel<<<blocks((long)L_CNT * HC_CNT * IN_CNT, 256), 256, 0, stream>>>(Wg, WgH, (long)L_CNT * HC_CNT * IN_CNT);
    f32_to_f16_kernel<<<blocks(128 * 384, 256), 256, 0, stream>>>(Wm1, Wm1H, 128 * 384);
    f32_to_f16_kernel<<<blocks(64 * 128, 256), 256, 0, stream>>>(Wm2, Wm2H, 64 * 128);

    for (int l = 0; l < L_CNT; ++l) {
        // h = x @ W^T  (WMMA)
        gat_gemm_kernel<<<dim3(N_CNT / 16, 4), 256, 0, stream>>>(
            xH, WgH + (size_t)l * HC_CNT * IN_CNT, hbuf);
        // per-(node,head) attention logits
        alpha_kernel<<<blocks((long)N_CNT * H_CNT * 32, 256), 256, 0, stream>>>(
            hbuf, att_s + (size_t)l * H_CNT * C_CNT, att_d + (size_t)l * H_CNT * C_CNT, aS, aD);
        // segment state init
        fill_u32_kernel<<<blocks((long)N_CNT * H_CNT, 256), 256, 0, stream>>>(mmax, ORD_NEG_INF, (long)N_CNT * H_CNT);
        fill_f32_kernel<<<blocks((long)N_CNT * H_CNT, 256), 256, 0, stream>>>(denom, 0.f, (long)N_CNT * H_CNT);
        fill_f32_kernel<<<blocks((long)N_CNT * HC_CNT, 256), 256, 0, stream>>>(outagg, 0.f, (long)N_CNT * HC_CNT);
        // softmax over incoming edges
        edge_max_kernel<<<blocks(ET_CNT, 256), 256, 0, stream>>>(ei, aS, aD, eexp, mmax);
        edge_exp_kernel<<<blocks(ET_CNT, 256), 256, 0, stream>>>(ei, eexp, mmax, denom);
        // weighted aggregation (atomic scatter, 2 edges/block, float4 gathers)
        edge_aggr_kernel<<<ET_CNT / 2, 256, 0, stream>>>(ei, eexp, denom, hbuf, outagg);
        // head mean + bias (+relu) -> next f16 input
        mean_kernel<<<blocks((long)N_CNT * IN_CNT, 256), 256, 0, stream>>>(
            outagg, bg + (size_t)l * C_CNT, (l == L_CNT - 1) ? hfinH : xH, (l < L_CNT - 1) ? 1 : 0);
    }

    // fused triplet MLP (all WMMA, LDS-resident intermediates)
    mlp_kernel<<<T_CNT / 16, 256, 0, stream>>>(hfinH, trips, Wm1H, bm1, Wm2H, bm2, Wm3, bm3, out);
}
